// SelfAttention2d_21191368638553
// MI455X (gfx1250) — matrix operand order repaired and checked by reading, not compiled, not run on hardware
//
#include <hip/hip_runtime.h>
#include <hip/hip_bf16.h>
#include <cstdint>
#include <cstddef>

#define NB 4
#define NC 64
#define NK 8
#define NN 4096

typedef __attribute__((ext_vector_type(16))) _Float16 v16h;
typedef __attribute__((ext_vector_type(8)))  _Float16 h8;
typedef __attribute__((ext_vector_type(8)))  float    v8f;

// monotonic float -> uint key (order-preserving, so integer atomicMax is a
// deterministic float max)
__device__ __forceinline__ unsigned fkey(float f) {
    unsigned u = __float_as_uint(f);
    return (u & 0x80000000u) ? ~u : (u | 0x80000000u);
}
__device__ __forceinline__ float funkey(unsigned k) {
    unsigned u = (k & 0x80000000u) ? (k ^ 0x80000000u) : ~k;
    return __uint_as_float(u);
}

#define WMMA16(A, B, C) \
    __builtin_amdgcn_wmma_f32_16x16x32_f16(false, (A), false, (B), (short)0, (C), false, false)

// ---------------------------------------------------------------- init
__global__ void sa2d_init(unsigned* __restrict__ Mf, unsigned* __restrict__ Mg,
                          float* __restrict__ Z, unsigned* __restrict__ z32) {
    int t = threadIdx.x;
    if (t < NB) { Mf[t] = 0u; Mg[t] = 0u; Z[t] = 0.f; }
    if (t < 8)  { z32[t] = 0u; }            // 32B zero line for idle-lane loads
}

// ------------------------------------------------- pass 1: projections
// fT: [B][N][8] f16 (A-operand columns). gB: [B][N][16] f16, B-ready layout:
// halves 0..7 = g[k][n], halves 8..15 = 0 (K zero-padding baked into data so
// the attention loop needs no EXEC branches or zero-fills).
__global__ __launch_bounds__(256) void sa2d_proj(
    const float* __restrict__ x,  const float* __restrict__ wf,
    const float* __restrict__ wg, const float* __restrict__ wh,
    _Float16* __restrict__ fT, _Float16* __restrict__ gB,
    _Float16* __restrict__ hh)
{
    __shared__ float xs[NC][65];            // 64c x 64n tile, padded
    const int b  = blockIdx.x >> 6;
    const int n0 = (blockIdx.x & 63) * 64;
    const int tid = threadIdx.x;

    for (int i = tid; i < NC * 64; i += 256) {
        int cc = i >> 6, nn = i & 63;
        xs[cc][nn] = x[((size_t)(b * NC + cc)) * NN + n0 + nn];
    }
    __syncthreads();

    const int nn  = tid & 63;
    const int grp = tid >> 6;               // 0..3

    // hh: each thread does 16 output channels for its column nn
    for (int j = 0; j < 16; ++j) {
        int c = grp * 16 + j;
        float acc = 0.f;
        #pragma unroll 8
        for (int cc = 0; cc < NC; ++cc) acc = fmaf(wh[c * NC + cc], xs[cc][nn], acc);
        hh[((size_t)(b * NC + c)) * NN + n0 + nn] = (_Float16)acc;
    }
    // f and g (8 rows each): 16 (k,which) pairs spread over the 4 groups
    for (int jj = 0; jj < 4; ++jj) {
        int sel = grp * 4 + jj;             // 0..15
        int k   = sel & 7;
        const float* w = (sel < 8) ? wf : wg;
        float acc = 0.f;
        #pragma unroll 8
        for (int cc = 0; cc < NC; ++cc) acc = fmaf(w[k * NC + cc], xs[cc][nn], acc);
        if (sel < 8) {
            fT[((size_t)b * NN + n0 + nn) * NK + k] = (_Float16)acc;
        } else {
            _Float16* dst = gB + ((size_t)b * NN + n0 + nn) * 16;
            dst[k]     = (_Float16)acc;     // K = k
            dst[8 + k] = (_Float16)0.f;     // K = 8..15 zero pad
        }
    }
}

// --------------------------- pass 2: Cauchy-Schwarz softmax shift bound
// Joint softmax is shift-invariant, so any M >= max(s) works exactly.
// max(s) <= max_n||f_n|| * max_m||g_m||; computing that is one cheap pass
// over the 256KB f/g arrays instead of 262K WMMAs over all N^2 scores.
__global__ __launch_bounds__(256) void sa2d_bound(
    const _Float16* __restrict__ fT, const _Float16* __restrict__ gB,
    unsigned* __restrict__ Mf, unsigned* __restrict__ Mg)
{
    const int b = blockIdx.x >> 4;
    const int n = (blockIdx.x & 15) * 256 + threadIdx.x;
    const int lane = threadIdx.x & 31;

    h8 fv = *(const h8*)(fT + ((size_t)b * NN + n) * NK);
    h8 gv = *(const h8*)(gB + ((size_t)b * NN + n) * 16);
    float fs = 0.f, gs = 0.f;
    #pragma unroll
    for (int i = 0; i < 8; ++i) {
        float a = (float)fv[i]; fs = fmaf(a, a, fs);
        float c = (float)gv[i]; gs = fmaf(c, c, gs);
    }
    fs = sqrtf(fs); gs = sqrtf(gs);
    #pragma unroll
    for (int off = 16; off >= 1; off >>= 1) {
        fs = fmaxf(fs, __shfl_xor(fs, off, 32));
        gs = fmaxf(gs, __shfl_xor(gs, off, 32));
    }
    if (lane == 0) {
        atomicMax(Mf + b, fkey(fs));
        atomicMax(Mg + b, fkey(gs));
    }
}

// ---------------- pass 3: fused exp + o = hh * p^T (flash-style, no s in HBM)
// m-range split 2-way into separate partial buffers (deterministic: no float
// atomics). gB B-operands are register double-buffered across iterations and
// hh A-operands are issued ahead of the s-WMMAs so load latency never sits on
// the WMMA critical path.
__global__ __launch_bounds__(256) void sa2d_attn(
    const _Float16* __restrict__ fT, const _Float16* __restrict__ gB,
    const _Float16* __restrict__ hh, const _Float16* __restrict__ z32,
    const unsigned* __restrict__ Mf, const unsigned* __restrict__ Mg,
    float* __restrict__ Zpart, float* __restrict__ oUn)
{
    // wave-private 16n x 32m f16 p-tile; rows padded to 40 halves (80B) so
    // ds_load_b128 is 16B-aligned and 16 row-reads hit distinct banks
    __shared__ __align__(16) _Float16 plds[8][16 * 40];

    const int lane  = threadIdx.x & 31;
    const int wave  = threadIdx.x >> 5;
    const int seg   = blockIdx.x >> 7;              // 2 m-segments
    const int r2    = blockIdx.x & 127;
    const int b     = r2 >> 5;
    const int ntile = (r2 & 31) * 8 + wave;
    const int n0    = ntile * 16;
    const int mstart = seg * (NN / 2);

    const float M = funkey(Mf[b]) * funkey(Mg[b]); // >= max(s), shift-exact

    v16h af{};
    if (lane < 16) {
        h8 v = *(const h8*)(fT + ((size_t)b * NN + n0 + lane) * NK);
        #pragma unroll
        for (int i = 0; i < 8; ++i) af[i] = v[i];
    }

    v8f acc[4];
    #pragma unroll
    for (int ct = 0; ct < 4; ++ct) acc[ct] = v8f{};
    float zloc = 0.f;

    _Float16* pw      = &plds[wave][0];
    const int row     = lane & 15;
    const int colbase = (lane < 16) ? 0 : 8;

    // branch-free B-operand pointers (idle upper lanes ride the zero line)
    const bool act = (lane < 16);
    const _Float16* gp =
        act ? (gB + ((size_t)b * NN + mstart + lane) * 16) : z32;
    const int gstep = act ? 512 : 0;
    const int goff2 = act ? 256 : 0;

    // hoisted hh A-operand row pointers (one per 16-c tile)
    const _Float16* hp[4];
    #pragma unroll
    for (int ct = 0; ct < 4; ++ct)
        hp[ct] = hh + ((size_t)(b * NC + ct * 16 + row)) * NN + mstart + (act ? 0 : 8);

    // prologue of the 2-deep software pipeline
    v16h nb0 = *(const v16h*)gp;
    v16h nb1 = *(const v16h*)(gp + goff2);
    gp += gstep;

    for (int mi = 0; mi < NN / 2; mi += 32) {
        const v16h b0 = nb0;
        const v16h b1 = nb1;
        // prefetch next chunk's B operands (final-iteration overread lands in
        // the adjacent hh workspace region: valid memory, values unused)
        nb0 = *(const v16h*)gp;
        nb1 = *(const v16h*)(gp + goff2);
        gp += gstep;
        // issue hh A-operand loads early; they are independent of the s path
        h8 lo[4], hi[4];
        #pragma unroll
        for (int ct = 0; ct < 4; ++ct) {
            lo[ct] = *(const h8*)hp[ct];         // K {0..7}  / {8..15}
            hi[ct] = *(const h8*)(hp[ct] + 16);  // K {16..23}/ {24..31}
            hp[ct] += 32;
        }

        v8f cs0{}, cs1{};
        cs0 = WMMA16(af, b0, cs0);
        cs1 = WMMA16(af, b1, cs1);

        // p = exp(s - M) -> wave-private LDS tile (C-layout scatter)
        #pragma unroll
        for (int r = 0; r < 8; ++r) {
            const int nl = act ? r : (8 + r);
            float p0 = __expf(cs0[r] - M);
            float p1 = __expf(cs1[r] - M);
            zloc += p0 + p1;
            pw[nl * 40 + row]      = (_Float16)p0;
            pw[nl * 40 + row + 16] = (_Float16)p1;
        }
        // B = p^T [32m x 16n]: lane = column n, halves = its 16 m-values
        const h8* pp = (const h8*)(pw + row * 40 + colbase);
        h8 q0 = pp[0], q1 = pp[2];
        v16h bp = __builtin_shufflevector(q0, q1, 0, 1, 2, 3, 4, 5, 6, 7,
                                          8, 9, 10, 11, 12, 13, 14, 15);

        // o[c, n-tile] += hh[c, m-chunk] * p^T, 4 c-tiles, f32 accumulate
        #pragma unroll
        for (int ct = 0; ct < 4; ++ct) {
            v16h ah = __builtin_shufflevector(lo[ct], hi[ct],
                                              0, 1, 2, 3, 4, 5, 6, 7,
                                              8, 9, 10, 11, 12, 13, 14, 15);
            acc[ct] = WMMA16(ah, bp, acc[ct]);
        }
    }

    float* oSeg = oUn + (size_t)seg * NB * NC * NN;
    #pragma unroll
    for (int ct = 0; ct < 4; ++ct)
        #pragma unroll
        for (int r = 0; r < 8; ++r) {
            const int c = ct * 16 + (act ? r : (8 + r));
            oSeg[((size_t)(b * NC + c)) * NN + n0 + row] = acc[ct][r];
        }

    #pragma unroll
    for (int off = 16; off >= 1; off >>= 1) zloc += __shfl_xor(zloc, off, 32);
    if (lane == 0) Zpart[(seg * NB + b) * 256 + ntile] = zloc;
}

// ------------------------------------------ deterministic Z reduction
__global__ void sa2d_zred(const float* __restrict__ Zpart, float* __restrict__ Z) {
    __shared__ float s[256];
    const int b = blockIdx.x, t = threadIdx.x;
    s[t] = Zpart[b * 256 + t] + Zpart[(NB + b) * 256 + t];
    __syncthreads();
    for (int off = 128; off > 0; off >>= 1) {
        if (t < off) s[t] += s[t + off];
        __syncthreads();
    }
    if (t == 0) Z[b] = s[0];
}

// ------------------------------------------------- pass 4: epilogue
__global__ void sa2d_fin(const float* __restrict__ oUn, const float* __restrict__ x,
                         const float* __restrict__ Z, const float* __restrict__ gamma,
                         float* __restrict__ out)
{
    const size_t i = (size_t)blockIdx.x * 256 + threadIdx.x;
    const int b = (int)(i >> 18);           // C*N = 2^18 elements per batch
    const float o = oUn[i] + oUn[i + (size_t)NB * NC * NN];
    out[i] = gamma[0] * o / Z[b] + x[i];
}

extern "C" void kernel_launch(void* const* d_in, const int* in_sizes, int n_in,
                              void* d_out, int out_size, void* d_ws, size_t ws_size,
                              hipStream_t stream) {
    (void)in_sizes; (void)n_in; (void)out_size; (void)ws_size;
    const float* x  = (const float*)d_in[0];
    const float* wf = (const float*)d_in[1];
    const float* wg = (const float*)d_in[2];
    const float* wh = (const float*)d_in[3];
    const float* gm = (const float*)d_in[4];
    float* out = (float*)d_out;

    char* ws = (char*)d_ws;
    const size_t oFT  = 0;
    const size_t oGB  = oFT + (size_t)NB * NN * NK * 2;        // 256 KB
    const size_t oHH  = oGB + (size_t)NB * NN * 16 * 2;        // 512 KB
    const size_t oOU  = oHH + (size_t)NB * NC * NN * 2;        // 2 MB
    const size_t oZP  = oOU + 2 * (size_t)NB * NC * NN * 4;    // 2 x 4 MB
    const size_t oZ   = oZP + 2 * (size_t)NB * 256 * 4;
    const size_t oMF  = oZ  + (size_t)NB * 4;
    const size_t oMG  = oMF + (size_t)NB * 4;
    const size_t oZ32 = ((oMG + (size_t)NB * 4 + 31) / 32) * 32;  // 32B zero line

    _Float16* fT  = (_Float16*)(ws + oFT);
    _Float16* gB  = (_Float16*)(ws + oGB);
    _Float16* hh  = (_Float16*)(ws + oHH);
    float*    oU  = (float*)(ws + oOU);
    float*    Zp  = (float*)(ws + oZP);
    float*    Z   = (float*)(ws + oZ);
    unsigned* Mf  = (unsigned*)(ws + oMF);
    unsigned* Mg  = (unsigned*)(ws + oMG);
    _Float16* z32 = (_Float16*)(ws + oZ32);

    sa2d_init<<<1, 64, 0, stream>>>(Mf, Mg, Z, (unsigned*)z32);
    sa2d_proj<<<NB * (NN / 64), 256, 0, stream>>>(x, wf, wg, wh, fT, gB, hh);
    sa2d_bound<<<NB * 16, 256, 0, stream>>>(fT, gB, Mf, Mg);
    sa2d_attn<<<2 * NB * 32, 256, 0, stream>>>(fT, gB, hh, z32, Mf, Mg, Zp, oU);
    sa2d_zred<<<NB, 256, 0, stream>>>(Zp, Z);
    sa2d_fin<<<(NB * NC * NN) / 256, 256, 0, stream>>>(oU, x, Z, gm, out);
}
